// RNNv2_3229815407361
// MI455X (gfx1250) — matrix operand order repaired and checked by reading, not compiled, not run on hardware
//
#include <hip/hip_runtime.h>
#include <cmath>

// ---------------------------------------------------------------------------
// RNN (4-layer, H=2048, S=2048, IN=1024) for MI455X / gfx1250.
// Strategy: bf16-convert all weights once (64MB -> L2-resident on 192MB L2),
// run the sequential recurrence in ONE persistent kernel with atomic grid
// barriers, and do every GEMV/GEMM through v_wmma_f32_16x16x32_bf16 with the
// h-vector broadcast across all 16 B columns (1KB of weights consumed per
// WMMA issue -> minimal instruction overhead for an L2-bandwidth-bound loop).
// ---------------------------------------------------------------------------

typedef __attribute__((ext_vector_type(16))) __bf16    bf16x16;
typedef __attribute__((ext_vector_type(8)))  float     f32x8;
typedef __attribute__((ext_vector_type(4)))  unsigned  u32x4;

union VA { bf16x16 v; u32x4 u[2]; };

constexpr int S  = 2048;   // sequence length
constexpr int IN = 1024;   // input / output feature size
constexpr int H  = 2048;   // hidden size
constexpr int L  = 4;      // layers

constexpr int NBLK = 32;               // persistent blocks (all co-resident)
constexpr int TPB  = 128;              // 4 waves (wave32)
constexpr int NWAVE = NBLK * (TPB / 32); // 128 waves == H/16 row-tiles

// ---- workspace layout (bytes) ---------------------------------------------
constexpr size_t SZ_WH   = (size_t)4 * H * H * 2;   // 32 MiB  bf16 W_h
constexpr size_t SZ_WHH  = (size_t)3 * H * H * 2;   // 24 MiB  bf16 W_hh
constexpr size_t SZ_WY   = (size_t)IN * H * 2;      //  4 MiB  bf16 W_y
constexpr size_t SZ_WX   = (size_t)H * IN * 2;      //  4 MiB  bf16 W_x
constexpr size_t SZ_XB   = (size_t)S * IN * 2;      //  4 MiB  bf16 x
constexpr size_t SZ_XIN  = (size_t)S * H * 4;       // 16 MiB  f32 xin
constexpr size_t SZ_HF   = (size_t)2 * L * H * 4;   // 64 KiB  f32 h (dbl buf)
constexpr size_t SZ_HB   = (size_t)2 * L * H * 2;   // 32 KiB  bf16 h (dbl buf)

constexpr size_t OFF_WH   = 0;
constexpr size_t OFF_WHH  = OFF_WH  + SZ_WH;
constexpr size_t OFF_WY   = OFF_WHH + SZ_WHH;
constexpr size_t OFF_WX   = OFF_WY  + SZ_WY;
constexpr size_t OFF_XB   = OFF_WX  + SZ_WX;
constexpr size_t OFF_XIN  = OFF_XB  + SZ_XB;
constexpr size_t OFF_HF   = OFF_XIN + SZ_XIN;
constexpr size_t OFF_HB   = OFF_HF  + SZ_HF;
constexpr size_t OFF_SYNC = OFF_HB  + SZ_HB;        // 2 ints: counter, gen

// ---------------------------------------------------------------------------
// f32 -> bf16 elementwise convert
// ---------------------------------------------------------------------------
__global__ void cvt_bf16_kernel(const float* __restrict__ in,
                                __bf16* __restrict__ out, int n) {
    int i = blockIdx.x * blockDim.x + threadIdx.x;
    if (i < n) out[i] = (__bf16)in[i];
}

// zero h state (both buffers, f32 + bf16) and barrier words
__global__ void init_state_kernel(float* hf, __bf16* hb, int* cnt, int* gen) {
    int i = blockIdx.x * blockDim.x + threadIdx.x;
    if (i < 2 * L * H) { hf[i] = 0.0f; hb[i] = (__bf16)0.0f; }
    if (i == 0) { *cnt = 0; *gen = 0; }
}

// ---------------------------------------------------------------------------
// Per-wave 16-row GEMV slab: acc += W[row0:row0+16, 0:K] @ hvec, with hvec
// broadcast into every B column (all 16 D columns identical).
// A layout (ISA 7.12.2, 16-bit 16x32): lane m=lane&15 holds row m, chunks at
// k offsets {0..7,16..23} (lanes 0-15) / {8..15,24..31} (lanes 16-31).
// B layout: same k chunks as A ({0..7,16..23} lanes 0-15 / {8..15,24..31} lanes 16-31).
// ---------------------------------------------------------------------------
__device__ __forceinline__
void wave_gemv(const __bf16* __restrict__ W, size_t ldw, int row0, int K,
               const __bf16* __restrict__ hvec, int lane, f32x8& acc)
{
    const int hi = lane >> 4;  // 0 | 1
    const __bf16* wp = W + (size_t)(row0 + (lane & 15)) * ldw + hi * 8;
    const __bf16* hp = hvec + hi * 8;
#pragma unroll 4
    for (int k = 0; k < K; k += 32) {
        VA a, b;
        a.u[0] = *(const u32x4*)(wp + k);
        a.u[1] = *(const u32x4*)(wp + k + 16);
        b.u[0] = *(const u32x4*)(hp + k);
        b.u[1] = *(const u32x4*)(hp + k + 16);
        acc = __builtin_amdgcn_wmma_f32_16x16x32_bf16(
                  false, a.v, false, b.v, (short)0, acc, false, false);
    }
}

// ---------------------------------------------------------------------------
// Device-wide barrier (persistent kernel; NBLK blocks all resident)
// ---------------------------------------------------------------------------
__device__ __forceinline__ void grid_sync(int* cnt, int* gen) {
    __syncthreads();
    if (threadIdx.x == 0) {
        __threadfence();
        int g = __atomic_load_n(gen, __ATOMIC_RELAXED);
        if (atomicAdd(cnt, 1) == NBLK - 1) {
            atomicExch(cnt, 0);
            __threadfence();
            atomicAdd(gen, 1);
        } else {
            while (__atomic_load_n(gen, __ATOMIC_RELAXED) == g)
                __builtin_amdgcn_s_sleep(2);
            __threadfence();
        }
    }
    __syncthreads();
}

// ---------------------------------------------------------------------------
// xin = x @ W_x.T  as bf16 WMMA GEMM, f32 accumulate. One wave per 16x16 tile.
// ---------------------------------------------------------------------------
__global__ void __launch_bounds__(TPB)
xin_gemm_kernel(const __bf16* __restrict__ xb, const __bf16* __restrict__ Wxb,
                float* __restrict__ xin)
{
    const int lane = threadIdx.x & 31;
    const int wid  = blockIdx.x * (TPB / 32) + (threadIdx.x >> 5);
    const int tm   = (wid / (H / 16)) * 16;   // timestep tile
    const int tn   = (wid % (H / 16)) * 16;   // hidden tile
    const int hi   = lane >> 4;

    const __bf16* ap = xb  + (size_t)(tm + (lane & 15)) * IN + hi * 8;
    const __bf16* bp = Wxb + (size_t)(tn + (lane & 15)) * IN + hi * 8;

    f32x8 acc = {};
#pragma unroll 4
    for (int k = 0; k < IN; k += 32) {
        VA a, b;
        a.u[0] = *(const u32x4*)(ap + k);
        a.u[1] = *(const u32x4*)(ap + k + 16);
        b.u[0] = *(const u32x4*)(bp + k);
        b.u[1] = *(const u32x4*)(bp + k + 16);
        acc = __builtin_amdgcn_wmma_f32_16x16x32_bf16(
                  false, a.v, false, b.v, (short)0, acc, false, false);
    }
    // D: N = lane&15, M = vgpr (+8 for upper half-wave)
    const int m0 = tm + hi * 8;
    const int n  = tn + (lane & 15);
#pragma unroll
    for (int r = 0; r < 8; ++r)
        xin[(size_t)(m0 + r) * H + n] = acc[r];
}

// ---------------------------------------------------------------------------
// Persistent recurrence: 2048 steps x 4 layer stages, 4 grid barriers/step.
// Wave w owns output rows [16w, 16w+16). W_y GEMV for step t-1 is overlapped
// into the (half-load) stage-0 slot of step t.
// ---------------------------------------------------------------------------
__global__ void __launch_bounds__(TPB)
rnn_recurrence_kernel(const __bf16* __restrict__ Whb,
                      const __bf16* __restrict__ Whhb,
                      const __bf16* __restrict__ Wyb,
                      const float*  __restrict__ xin,
                      const float*  __restrict__ Bh,
                      const float*  __restrict__ By,
                      float* hf, __bf16* hb,
                      float* __restrict__ ys, float* __restrict__ hfin,
                      int* cnt, int* gen)
{
    const int lane = threadIdx.x & 31;
    const int w    = blockIdx.x * (TPB / 32) + (threadIdx.x >> 5);
    const int row0 = w * 16;
    const int hi   = lane >> 4;

#pragma unroll 1
    for (int t = 0; t < S; ++t) {
        const int cur = t & 1, nxt = cur ^ 1;

        // ---- stage 0: h0' = tanh(W_h0 @ h0 + xin[t] + Bh0) ----
        {
            f32x8 acc = {};
            wave_gemv(Whb, H, row0, H, hb + (size_t)(cur * L + 0) * H, lane, acc);
            if (lane == 0 || lane == 16) {
                const int mb = row0 + hi * 8;
#pragma unroll
                for (int r = 0; r < 8; ++r) {
                    float v = tanhf(acc[r] + xin[(size_t)t * H + mb + r] + Bh[mb + r]);
                    hf[(size_t)(nxt * L + 0) * H + mb + r] = v;
                    hb[(size_t)(nxt * L + 0) * H + mb + r] = (__bf16)v;
                }
            }
            // overlapped: y[t-1] = W_y @ h3(t-1) + By  (64 waves)
            if (t > 0 && w < IN / 16) {
                f32x8 ya = {};
                wave_gemv(Wyb, H, row0, H, hb + (size_t)(cur * L + 3) * H, lane, ya);
                if (lane == 0 || lane == 16) {
                    const int mb = row0 + hi * 8;
#pragma unroll
                    for (int r = 0; r < 8; ++r)
                        ys[(size_t)(t - 1) * IN + mb + r] = ya[r] + By[mb + r];
                }
            }
        }
        grid_sync(cnt, gen);

        // ---- stages 1..3: hl' = tanh(W_hl @ hl + W_hh(l-1) @ h(l-1)' + Bhl)
#pragma unroll 1
        for (int l = 1; l < L; ++l) {
            f32x8 acc = {};
            wave_gemv(Whb  + (size_t)l * H * H,       H, row0, H,
                      hb + (size_t)(cur * L + l) * H,       lane, acc);
            wave_gemv(Whhb + (size_t)(l - 1) * H * H, H, row0, H,
                      hb + (size_t)(nxt * L + (l - 1)) * H, lane, acc);
            if (lane == 0 || lane == 16) {
                const int mb = row0 + hi * 8;
#pragma unroll
                for (int r = 0; r < 8; ++r) {
                    float v = tanhf(acc[r] + Bh[l * H + mb + r]);
                    hf[(size_t)(nxt * L + l) * H + mb + r] = v;
                    hb[(size_t)(nxt * L + l) * H + mb + r] = (__bf16)v;
                }
            }
            grid_sync(cnt, gen);
        }
    }

    // ---- tail: y[S-1] and h_final (final state lives in buffer 0: S even)
    if (w < IN / 16) {
        f32x8 ya = {};
        wave_gemv(Wyb, H, row0, H, hb + (size_t)(0 * L + 3) * H, lane, ya);
        if (lane == 0 || lane == 16) {
            const int mb = row0 + hi * 8;
#pragma unroll
            for (int r = 0; r < 8; ++r)
                ys[(size_t)(S - 1) * IN + mb + r] = ya[r] + By[mb + r];
        }
    }
    for (int i = blockIdx.x * TPB + threadIdx.x; i < L * H; i += NBLK * TPB)
        hfin[i] = hf[i];   // buffer 0, [L][H] flat
}

// ---------------------------------------------------------------------------
extern "C" void kernel_launch(void* const* d_in, const int* in_sizes, int n_in,
                              void* d_out, int out_size, void* d_ws, size_t ws_size,
                              hipStream_t stream) {
    const float* x   = (const float*)d_in[0];  // [S, IN]
    const float* W_h = (const float*)d_in[1];  // [L, H, H]
    const float* W_hh= (const float*)d_in[2];  // [L-1, H, H]
    const float* W_x = (const float*)d_in[3];  // [H, IN]
    const float* W_y = (const float*)d_in[4];  // [IN, H]
    const float* B_h = (const float*)d_in[5];  // [L, H]
    const float* B_y = (const float*)d_in[6];  // [IN]

    char* ws = (char*)d_ws;
    __bf16* Whb  = (__bf16*)(ws + OFF_WH);
    __bf16* Whhb = (__bf16*)(ws + OFF_WHH);
    __bf16* Wyb  = (__bf16*)(ws + OFF_WY);
    __bf16* Wxb  = (__bf16*)(ws + OFF_WX);
    __bf16* xb   = (__bf16*)(ws + OFF_XB);
    float*  xin  = (float* )(ws + OFF_XIN);
    float*  hf   = (float* )(ws + OFF_HF);
    __bf16* hb   = (__bf16*)(ws + OFF_HB);
    int*    cnt  = (int*   )(ws + OFF_SYNC);
    int*    gen  = cnt + 1;

    float* ys   = (float*)d_out;              // [S, IN]
    float* hfin = (float*)d_out + (size_t)S * IN;  // [L, H]

    auto cvt = [&](const float* src, __bf16* dst, int n) {
        cvt_bf16_kernel<<<(n + 255) / 256, 256, 0, stream>>>(src, dst, n);
    };
    cvt(W_h,  Whb,  4 * H * H);
    cvt(W_hh, Whhb, 3 * H * H);
    cvt(W_y,  Wyb,  IN * H);
    cvt(W_x,  Wxb,  H * IN);
    cvt(x,    xb,   S * IN);

    init_state_kernel<<<(2 * L * H + 255) / 256, 256, 0, stream>>>(hf, hb, cnt, gen);

    // xin GEMM: (S/16)*(H/16) = 16384 wave-tiles, 4 waves/block
    const int tiles = (S / 16) * (H / 16);
    xin_gemm_kernel<<<tiles / (TPB / 32), TPB, 0, stream>>>(xb, Wxb, xin);

    // persistent recurrence
    rnn_recurrence_kernel<<<NBLK, TPB, 0, stream>>>(
        Whb, Whhb, Wyb, xin, B_h, B_y, hf, hb, ys, hfin, cnt, gen);
}
